// GAT_1322849927892
// MI455X (gfx1250) — matrix operand order repaired and checked by reading, not compiled, not run on hardware
//
#include <hip/hip_runtime.h>
#include <hip/hip_bf16.h>

// ---------------- problem constants (match reference) ----------------
constexpr int Nn   = 16384;   // nodes
constexpr int Dd   = 1024;    // input dim
constexpr int Ee   = 65536;   // edges (before self loops)
constexpr int ETOT = Ee + Nn; // 81920 edges incl. self loops
constexpr int H1c  = 8;       // heads layer 1
constexpr int HIDc = 512;     // channels per head
constexpr int F1   = H1c * HIDc; // 4096
// layer 2: 4096 -> 2, 1 head

// ---------------- vector types for WMMA ----------------
typedef __attribute__((ext_vector_type(16))) __bf16 v16bf;
typedef __attribute__((ext_vector_type(8)))  __bf16 v8bf;
typedef __attribute__((ext_vector_type(8)))  float  v8f;

// ---------------- helpers ----------------
__device__ __forceinline__ int edge_src(const int* ei, int e) {
  return (e < Ee) ? ei[e] : (e - Ee);
}
__device__ __forceinline__ int edge_dst(const int* ei, int e) {
  return (e < Ee) ? ei[Ee + e] : (e - Ee);
}
// monotone float<->uint mapping for atomicMax on floats
__device__ __forceinline__ unsigned encf(float f) {
  int i = __float_as_int(f);
  return (i >= 0) ? ((unsigned)i | 0x80000000u) : ~(unsigned)i;
}
__device__ __forceinline__ float decf(unsigned u) {
  int i = (u & 0x80000000u) ? (int)(u & 0x7fffffffu) : (int)(~u);
  return __int_as_float(i);
}
#define ENC_NEG_FLT_MAX 0x00800000u   // encf(-FLT_MAX)

// ---------------- generic small kernels ----------------
__global__ void f2bf_kernel(const float* __restrict__ in, __bf16* __restrict__ out, int n) {
  int t = blockIdx.x * 256 + threadIdx.x;
  if (t < n) out[t] = (__bf16)in[t];
}
__global__ void fill_u32_kernel(unsigned* __restrict__ p, unsigned v, int n) {
  int t = blockIdx.x * 256 + threadIdx.x;
  if (t < n) p[t] = v;
}
__global__ void decode_max_kernel(const unsigned* __restrict__ enc, float* __restrict__ dec, int n) {
  int t = blockIdx.x * 256 + threadIdx.x;
  if (t < n) {
    unsigned u = enc[t];
    dec[t] = (u == ENC_NEG_FLT_MAX) ? 0.0f : decf(u);  // isfinite guard
  }
}

// fold attention vectors into weights:  V[d,h] = sum_c W[d, h*512+c] * att[h,c]
__global__ void fold_att_kernel(const float* __restrict__ W1s, const float* __restrict__ a1s,
                                const float* __restrict__ W1d, const float* __restrict__ a1d,
                                float* __restrict__ Vs, float* __restrict__ Vd) {
  int t = blockIdx.x * 256 + threadIdx.x;   // Dd*H1c = 8192
  if (t >= Dd * H1c) return;
  int d = t >> 3, h = t & 7;
  const float* ws = W1s + (size_t)d * F1 + h * HIDc;
  const float* wd = W1d + (size_t)d * F1 + h * HIDc;
  const float* as = a1s + h * HIDc;
  const float* ad = a1d + h * HIDc;
  float s = 0.f, dd = 0.f;
  for (int c = 0; c < HIDc; ++c) { s += ws[c] * as[c]; dd += wd[c] * ad[c]; }
  Vs[d * H1c + h] = s;
  Vd[d * H1c + h] = dd;
}

// al_s[n,h] = x[n,:] . Vs[:,h] ; al_d likewise (x read in fp32, L2-resident V)
__global__ void al_kernel(const float* __restrict__ x, const float* __restrict__ Vs,
                          const float* __restrict__ Vd,
                          float* __restrict__ al_s, float* __restrict__ al_d) {
  int t = blockIdx.x * 256 + threadIdx.x;   // Nn*H1c
  if (t >= Nn * H1c) return;
  int n = t >> 3, h = t & 7;
  const float* xr = x + (size_t)n * Dd;
  float s = 0.f, d = 0.f;
  for (int k = 0; k < Dd; ++k) {
    float xv = xr[k];
    s += xv * Vs[k * H1c + h];
    d += xv * Vd[k * H1c + h];
  }
  al_s[t] = s;
  al_d[t] = d;
}

// ---------------- the WMMA GEMM: C[M,N] = A[M,K] * B[K,N], bf16 in, bf16 out ----
// BM=128 BN=128 BK=32, 256 threads = 8 waves (4 x 2), wave tile 32x64.
// Double-buffered LDS (ping-pong): next tile's global loads are issued into
// registers before compute on the current tile; one barrier per k-step.
__global__ __launch_bounds__(256) void gemm_bf16_wmma(const __bf16* __restrict__ A,
                                                      const __bf16* __restrict__ B,
                                                      __bf16* __restrict__ C,
                                                      int M, int N, int K) {
  constexpr int BM = 128, BN = 128, BK = 32;
  __shared__ __align__(16) __bf16 As[2][BM * BK];     // [buf][row][k]
  __shared__ __align__(16) __bf16 Bt[2][BN * BK];     // [buf][col][k] (transposed)

  const int tid  = threadIdx.x;
  const int wave = tid >> 5, lane = tid & 31;
  const int wm = wave & 3, wn = wave >> 2;            // 4 x 2 wave grid
  const int half = lane >> 4, l16 = lane & 15;

  const int rowBlk = blockIdx.x * BM;
  const int colBlk = blockIdx.y * BN;

  // per-thread tile-chunk coordinates (two 8-element chunks per tile)
  int aRow[2], aCol[2], bKk[2], bN[2];
#pragma unroll
  for (int rep = 0; rep < 2; ++rep) {
    int chunk = tid + rep * 256;        // 0..511 chunks of 8 bf16
    aRow[rep] = chunk >> 2;             // /4   -> 0..127
    aCol[rep] = (chunk & 3) * 8;        // 0..24
    bKk[rep]  = chunk >> 4;             // /16  -> 0..31
    bN[rep]   = (chunk & 15) * 8;       // 0..120
  }

  v8f acc[2][4];
  const v8f vzero = {0.f, 0.f, 0.f, 0.f, 0.f, 0.f, 0.f, 0.f};
#pragma unroll
  for (int i = 0; i < 2; ++i)
#pragma unroll
    for (int j = 0; j < 4; ++j) acc[i][j] = vzero;

  // ---- preload k-tile 0 into buffer 0 ----
  v8bf rA[2], rB[2];
#pragma unroll
  for (int rep = 0; rep < 2; ++rep) {
    rA[rep] = *(const v8bf*)(A + (size_t)(rowBlk + aRow[rep]) * K + aCol[rep]);
    rB[rep] = *(const v8bf*)(B + (size_t)bKk[rep] * N + colBlk + bN[rep]);
  }
#pragma unroll
  for (int rep = 0; rep < 2; ++rep) {
    *(v8bf*)(&As[0][aRow[rep] * BK + aCol[rep]]) = rA[rep];
#pragma unroll
    for (int i = 0; i < 8; ++i) Bt[0][(bN[rep] + i) * BK + bKk[rep]] = rB[rep][i];
  }
  __syncthreads();

  const int KT = K / BK;
  for (int kt = 0; kt < KT; ++kt) {
    const int cur = kt & 1, nxt = cur ^ 1;
    const bool hasNext = (kt + 1 < KT);

    // ---- issue next tile's global loads before compute (latency hiding) ----
    if (hasNext) {
      const int k1 = (kt + 1) * BK;
      if (kt + 2 < KT) {   // warm WGP$ two tiles ahead (global_prefetch_b8)
        __builtin_prefetch(A + (size_t)(rowBlk + aRow[0]) * K + (kt + 2) * BK, 0, 1);
        __builtin_prefetch(B + (size_t)((kt + 2) * BK + bKk[0]) * N + colBlk, 0, 1);
      }
#pragma unroll
      for (int rep = 0; rep < 2; ++rep) {
        rA[rep] = *(const v8bf*)(A + (size_t)(rowBlk + aRow[rep]) * K + k1 + aCol[rep]);
        rB[rep] = *(const v8bf*)(B + (size_t)(k1 + bKk[rep]) * N + colBlk + bN[rep]);
      }
    }

    // ---- build fragments per CDNA5 16-bit WMMA layouts ----
    // A (16x32): lane l16 = row M, half selects K groups {0..7,16..23} / {8..15,24..31}
    v16bf afrag[2];
#pragma unroll
    for (int mt = 0; mt < 2; ++mt) {
      const __bf16* ap = &As[cur][(size_t)(wm * 32 + mt * 16 + l16) * BK];
      v8bf lo = *(const v8bf*)(ap + half * 8);
      v8bf hi = *(const v8bf*)(ap + 16 + half * 8);
#pragma unroll
      for (int i = 0; i < 8; ++i) { afrag[mt][i] = lo[i]; afrag[mt][i + 8] = hi[i]; }
    }
    // B (32x16): lane l16 = column N, half selects K groups {0..7,16..23} / {8..15,24..31}
    v16bf bfrag[4];
#pragma unroll
    for (int nt = 0; nt < 4; ++nt) {
      const __bf16* bp = &Bt[cur][(size_t)(wn * 64 + nt * 16 + l16) * BK + half * 8];
      v8bf lo = *(const v8bf*)(bp);
      v8bf hi = *(const v8bf*)(bp + 16);
#pragma unroll
      for (int i = 0; i < 8; ++i) { bfrag[nt][i] = lo[i]; bfrag[nt][i + 8] = hi[i]; }
    }

#pragma unroll
    for (int mt = 0; mt < 2; ++mt)
#pragma unroll
      for (int nt = 0; nt < 4; ++nt)
        acc[mt][nt] = __builtin_amdgcn_wmma_f32_16x16x32_bf16(
            false, afrag[mt], false, bfrag[nt], (short)0, acc[mt][nt], false, false);

    // ---- stage next tile into the other LDS buffer; single barrier ----
    if (hasNext) {
#pragma unroll
      for (int rep = 0; rep < 2; ++rep) {
        *(v8bf*)(&As[nxt][aRow[rep] * BK + aCol[rep]]) = rA[rep];
#pragma unroll
        for (int i = 0; i < 8; ++i) Bt[nxt][(bN[rep] + i) * BK + bKk[rep]] = rB[rep][i];
      }
      __syncthreads();
    }
  }

  // ---- epilogue: C/D layout -> element r of v8f is row (r + 8*half), col l16 ----
#pragma unroll
  for (int mt = 0; mt < 2; ++mt)
#pragma unroll
    for (int nt = 0; nt < 4; ++nt) {
      int row0 = rowBlk + wm * 32 + mt * 16 + half * 8;
      int col  = colBlk + wn * 64 + nt * 16 + l16;
#pragma unroll
      for (int r = 0; r < 8; ++r)
        C[(size_t)(row0 + r) * N + col] = (__bf16)acc[mt][nt][r];
    }
}

// ---------------- layer-1 edge softmax ----------------
__global__ void edge1_logit_kernel(const int* __restrict__ ei,
                                   const float* __restrict__ al_s, const float* __restrict__ al_d,
                                   float* __restrict__ lg, unsigned* __restrict__ menc) {
  int t = blockIdx.x * 256 + threadIdx.x;   // ETOT * H1c
  if (t >= ETOT * H1c) return;
  int e = t >> 3, h = t & 7;
  int s = edge_src(ei, e), d = edge_dst(ei, e);
  float v = al_s[s * H1c + h] + al_d[d * H1c + h];
  v = (v > 0.f) ? v : 0.2f * v;                 // leaky_relu(0.2)
  lg[t] = v;
  atomicMax(&menc[d * H1c + h], encf(v));
}
__global__ void edge1_exp_kernel(const int* __restrict__ ei,
                                 const float* __restrict__ lg, const float* __restrict__ mdec,
                                 float* __restrict__ ev, float* __restrict__ denom) {
  int t = blockIdx.x * 256 + threadIdx.x;
  if (t >= ETOT * H1c) return;
  int e = t >> 3, h = t & 7;
  int d = edge_dst(ei, e);
  float x = expf(lg[t] - mdec[d * H1c + h]);
  ev[t] = x;
  atomicAdd(&denom[d * H1c + h], x);
}
// one block per edge: out[dst, :] += alpha[h] * hs[src, :]
__global__ __launch_bounds__(256) void scatter1_kernel(const int* __restrict__ ei,
                                                       const float* __restrict__ ev,
                                                       const float* __restrict__ denom,
                                                       const __bf16* __restrict__ hsb,
                                                       float* __restrict__ hacc) {
  int e = blockIdx.x;
  int s = edge_src(ei, e), d = edge_dst(ei, e);
  __shared__ float alpha[H1c];
  if (threadIdx.x < H1c)
    alpha[threadIdx.x] = ev[e * H1c + threadIdx.x] / (denom[d * H1c + threadIdx.x] + 1e-16f);
  __syncthreads();
  const __bf16* src = hsb  + (size_t)s * F1;
  float*        dst = hacc + (size_t)d * F1;
  for (int j = threadIdx.x; j < F1; j += 256)
    atomicAdd(dst + j, alpha[j >> 9] * (float)src[j]);
}

// ---------------- layer 2 ----------------
__global__ void fold2_kernel(const float* __restrict__ W2d, const float* __restrict__ a2d,
                             float* __restrict__ ud) {
  int j = blockIdx.x * 256 + threadIdx.x;
  if (j < F1) ud[j] = W2d[j * 2] * a2d[0] + W2d[j * 2 + 1] * a2d[1];
}
// wave-per-node GEMV: hs2[n,0..1], al_s2[n], al_d2[n] from (h+b1)
__global__ __launch_bounds__(256) void l2_gemv_kernel(const float* __restrict__ hacc,
                                                      const float* __restrict__ b1,
                                                      const float* __restrict__ W2s,
                                                      const float* __restrict__ ud,
                                                      const float* __restrict__ a2s,
                                                      float* __restrict__ hs2,
                                                      float* __restrict__ als2,
                                                      float* __restrict__ ald2) {
  int wave = threadIdx.x >> 5, lane = threadIdx.x & 31;
  int n = blockIdx.x * 8 + wave;
  const float* hr = hacc + (size_t)n * F1;
  float s0 = 0.f, s1 = 0.f, ad = 0.f;
  for (int j = lane; j < F1; j += 32) {
    float hv = hr[j] + b1[j];
    s0 += hv * W2s[j * 2];
    s1 += hv * W2s[j * 2 + 1];
    ad += hv * ud[j];
  }
#pragma unroll
  for (int m = 16; m >= 1; m >>= 1) {
    s0 += __shfl_xor(s0, m, 32);
    s1 += __shfl_xor(s1, m, 32);
    ad += __shfl_xor(ad, m, 32);
  }
  if (lane == 0) {
    hs2[n * 2]     = s0;
    hs2[n * 2 + 1] = s1;
    ald2[n]        = ad;
    als2[n]        = s0 * a2s[0] + s1 * a2s[1];
  }
}
__global__ void edge2_logit_kernel(const int* __restrict__ ei,
                                   const float* __restrict__ als2, const float* __restrict__ ald2,
                                   float* __restrict__ lg2, unsigned* __restrict__ m2e) {
  int e = blockIdx.x * 256 + threadIdx.x;
  if (e >= ETOT) return;
  int s = edge_src(ei, e), d = edge_dst(ei, e);
  float v = als2[s] + ald2[d];
  v = (v > 0.f) ? v : 0.2f * v;
  lg2[e] = v;
  atomicMax(&m2e[d], encf(v));
}
__global__ void edge2_exp_kernel(const int* __restrict__ ei,
                                 const float* __restrict__ lg2, const float* __restrict__ m2d,
                                 float* __restrict__ ev2, float* __restrict__ den2) {
  int e = blockIdx.x * 256 + threadIdx.x;
  if (e >= ETOT) return;
  int d = edge_dst(ei, e);
  float x = expf(lg2[e] - m2d[d]);
  ev2[e] = x;
  atomicAdd(&den2[d], x);
}
__global__ void scatter2_kernel(const int* __restrict__ ei,
                                const float* __restrict__ ev2, const float* __restrict__ den2,
                                const float* __restrict__ hs2, float* __restrict__ o2) {
  int e = blockIdx.x * 256 + threadIdx.x;
  if (e >= ETOT) return;
  int s = edge_src(ei, e), d = edge_dst(ei, e);
  float a = ev2[e] / (den2[d] + 1e-16f);
  atomicAdd(&o2[d * 2],     a * hs2[s * 2]);
  atomicAdd(&o2[d * 2 + 1], a * hs2[s * 2 + 1]);
}
__global__ void final_out_kernel(const float* __restrict__ o2, const float* __restrict__ b2,
                                 float* __restrict__ out) {
  int t = blockIdx.x * 256 + threadIdx.x;
  if (t < Nn * 2) out[t] = o2[t] + b2[t & 1];
}

// ---------------- host launcher ----------------
extern "C" void kernel_launch(void* const* d_in, const int* in_sizes, int n_in,
                              void* d_out, int out_size, void* d_ws, size_t ws_size,
                              hipStream_t stream) {
  (void)in_sizes; (void)n_in; (void)out_size; (void)ws_size;
  const float* x   = (const float*)d_in[0];
  const int*   ei  = (const int*)d_in[1];
  const float* W1s = (const float*)d_in[2];
  const float* W1d = (const float*)d_in[3];
  const float* a1s = (const float*)d_in[4];
  const float* a1d = (const float*)d_in[5];
  const float* b1  = (const float*)d_in[6];
  const float* W2s = (const float*)d_in[7];
  const float* W2d = (const float*)d_in[8];
  const float* a2s = (const float*)d_in[9];
  const float* a2d = (const float*)d_in[10];
  const float* b2  = (const float*)d_in[11];
  float* out = (float*)d_out;

  // ---- carve workspace (256B aligned slabs) ----
  char* base = (char*)d_ws;
  size_t off = 0;
  auto carve = [&](size_t bytes) -> char* {
    char* p = base + off;
    off = (off + bytes + 255) & ~(size_t)255;
    return p;
  };
  __bf16*  xb   = (__bf16*) carve((size_t)Nn * Dd * 2);
  __bf16*  w1b  = (__bf16*) carve((size_t)Dd * F1 * 2);
  __bf16*  hsb  = (__bf16*) carve((size_t)Nn * F1 * 2);
  float*   Vs   = (float*)  carve((size_t)Dd * H1c * 4);
  float*   Vd   = (float*)  carve((size_t)Dd * H1c * 4);
  float*   al_s = (float*)  carve((size_t)Nn * H1c * 4);
  float*   al_d = (float*)  carve((size_t)Nn * H1c * 4);
  unsigned* menc= (unsigned*)carve((size_t)Nn * H1c * 4);
  float*   mdec = (float*)  carve((size_t)Nn * H1c * 4);
  float*   denom= (float*)  carve((size_t)Nn * H1c * 4);
  float*   lg1  = (float*)  carve((size_t)ETOT * H1c * 4);
  float*   ev1  = (float*)  carve((size_t)ETOT * H1c * 4);
  float*   hacc = (float*)  carve((size_t)Nn * F1 * 4);
  float*   ud   = (float*)  carve((size_t)F1 * 4);
  float*   hs2  = (float*)  carve((size_t)Nn * 2 * 4);
  float*   als2 = (float*)  carve((size_t)Nn * 4);
  float*   ald2 = (float*)  carve((size_t)Nn * 4);
  unsigned* m2e = (unsigned*)carve((size_t)Nn * 4);
  float*   m2d  = (float*)  carve((size_t)Nn * 4);
  float*   den2 = (float*)  carve((size_t)Nn * 4);
  float*   lg2  = (float*)  carve((size_t)ETOT * 4);
  float*   ev2  = (float*)  carve((size_t)ETOT * 4);
  float*   o2   = (float*)  carve((size_t)Nn * 2 * 4);

  // ---- init accumulators (deterministic per call) ----
  hipMemsetAsync(denom, 0, (size_t)Nn * H1c * 4, stream);
  hipMemsetAsync(hacc,  0, (size_t)Nn * F1 * 4, stream);
  hipMemsetAsync(den2,  0, (size_t)Nn * 4, stream);
  hipMemsetAsync(o2,    0, (size_t)Nn * 2 * 4, stream);
  fill_u32_kernel<<<(Nn * H1c + 255) / 256, 256, 0, stream>>>(menc, ENC_NEG_FLT_MAX, Nn * H1c);
  fill_u32_kernel<<<(Nn + 255) / 256, 256, 0, stream>>>(m2e, ENC_NEG_FLT_MAX, Nn);

  // ---- precision converts + attention folding ----
  f2bf_kernel<<<(Nn * Dd + 255) / 256, 256, 0, stream>>>(x, xb, Nn * Dd);
  f2bf_kernel<<<(Dd * F1 + 255) / 256, 256, 0, stream>>>(W1s, w1b, Dd * F1);
  fold_att_kernel<<<(Dd * H1c + 255) / 256, 256, 0, stream>>>(W1s, a1s, W1d, a1d, Vs, Vd);

  // ---- layer 1: hs = x @ W1_src via WMMA ----
  dim3 ggrid(Nn / 128, F1 / 128);
  gemm_bf16_wmma<<<ggrid, 256, 0, stream>>>(xb, w1b, hsb, Nn, F1, Dd);

  // ---- layer 1: attention coefficients + segment softmax ----
  al_kernel<<<(Nn * H1c + 255) / 256, 256, 0, stream>>>(x, Vs, Vd, al_s, al_d);
  edge1_logit_kernel<<<(ETOT * H1c + 255) / 256, 256, 0, stream>>>(ei, al_s, al_d, lg1, menc);
  decode_max_kernel<<<(Nn * H1c + 255) / 256, 256, 0, stream>>>(menc, mdec, Nn * H1c);
  edge1_exp_kernel<<<(ETOT * H1c + 255) / 256, 256, 0, stream>>>(ei, lg1, mdec, ev1, denom);
  scatter1_kernel<<<ETOT, 256, 0, stream>>>(ei, ev1, denom, hsb, hacc);

  // ---- layer 2 ----
  fold2_kernel<<<(F1 + 255) / 256, 256, 0, stream>>>(W2d, a2d, ud);
  l2_gemv_kernel<<<Nn / 8, 256, 0, stream>>>(hacc, b1, W2s, ud, a2s, hs2, als2, ald2);
  edge2_logit_kernel<<<(ETOT + 255) / 256, 256, 0, stream>>>(ei, als2, ald2, lg2, m2e);
  decode_max_kernel<<<(Nn + 255) / 256, 256, 0, stream>>>(m2e, m2d, Nn);
  edge2_exp_kernel<<<(ETOT + 255) / 256, 256, 0, stream>>>(ei, lg2, m2d, ev2, den2);
  scatter2_kernel<<<(ETOT + 255) / 256, 256, 0, stream>>>(ei, ev2, den2, hs2, o2);
  final_out_kernel<<<(Nn * 2 + 255) / 256, 256, 0, stream>>>(o2, b2, out);
}
